// PixelCNNDistribution_37735582663393
// MI455X (gfx1250) — matrix operand order repaired and checked by reading, not compiled, not run on hardware
//
#include <hip/hip_runtime.h>
#include <hip/hip_bf16.h>

typedef __attribute__((ext_vector_type(16))) _Float16 v16h;
typedef __attribute__((ext_vector_type(8)))  _Float16 v8h;
typedef __attribute__((ext_vector_type(8)))  float    v8f;

#define NPIX 65536   // 16 * 64 * 64
#define KBLK 800     // 5 live causal taps * 160 channels (tap-major: k = tap*160 + cin)
#define LDA  48      // padded LDS leading dim (halfs) for A tile rows
#define LDB  48      // padded LDS leading dim (halfs) for B tile cols (col-major)

// ---------------------------------------------------------------- helpers
__device__ __forceinline__ v8f wmma_f16(v16h a, v16h b, v8f c) {
  return __builtin_amdgcn_wmma_f32_16x16x32_f16(
      /*neg_a=*/false, a, /*neg_b=*/false, b,
      /*c_mod=*/(short)0, c, /*reuse_a=*/false, /*reuse_b=*/false);
}

__device__ __forceinline__ v16h cat16(v8h lo, v8h hi) {
  return __builtin_shufflevector(lo, hi, 0, 1, 2, 3, 4, 5, 6, 7,
                                 8, 9, 10, 11, 12, 13, 14, 15);
}

// Async copy 16B/lane from global to LDS (GLOBAL_LOAD_ASYNC_TO_LDS_B128, ASYNCcnt).
__device__ __forceinline__ void async_copy_b128(unsigned lds_off, const void* gptr) {
  asm volatile("global_load_async_to_lds_b128 %0, %1, off"
               :
               : "v"(lds_off), "v"((unsigned long long)(uintptr_t)gptr)
               : "memory");
}
__device__ __forceinline__ void wait_async0() {
  asm volatile("s_wait_asynccnt 0" ::: "memory");
}

// A fragment: 16x32 f16 tile at row offset mo from LDS [64][LDA] row-major.
// ISA: lanes 0-15 M=lane, K {0..7,16..23}; lanes 16-31 K {8..15,24..31}.
__device__ __forceinline__ v16h load_frag_A(const _Float16* A, int lane, int mo) {
  int row   = mo + (lane & 15);
  int kbase = (lane >> 4) * 8;
  const _Float16* p = A + row * LDA + kbase;
  v8h lo = *(const v8h*)p;        // K = kbase .. kbase+7
  v8h hi = *(const v8h*)(p + 16); // K = 16+kbase .. 16+kbase+7
  return cat16(lo, hi);
}

// B fragment: 32x16 f16 tile at col offset no from LDS stored COLUMN-major
// [128][LDB] (col, k).  col = lane&15, K {0..7,16..23} lanes 0-15; {8..15,24..31} lanes 16-31.
__device__ __forceinline__ v16h load_frag_B(const _Float16* B, int lane, int no) {
  int col = no + (lane & 15);
  int kb  = (lane >> 4) * 8;
  const _Float16* p = B + col * LDB + kb;
  v8h lo = *(const v8h*)p;
  v8h hi = *(const v8h*)(p + 16);
  return cat16(lo, hi);
}

__device__ __forceinline__ float softplusf_(float x) {
  return (x > 20.f) ? x : log1pf(__expf(x));
}
__device__ __forceinline__ float sigmoidf_(float x) {
  return 1.f / (1.f + __expf(-x));
}

// ---------------------------------------------------------------- prep kernels
__global__ void zero_out_kernel(float* out) {
  if (threadIdx.x < 16) out[threadIdx.x] = 0.f;
}

// x = samples*2-1, relayout NCHW -> [C][B*H*W]
__global__ void prep_x_kernel(const float* __restrict__ samples, float* __restrict__ x32) {
  int i = blockIdx.x * 256 + threadIdx.x;
  if (i >= 3 * NPIX) return;
  int c = i >> 16;
  int n = i & (NPIX - 1);
  int b = n >> 12;
  int p = n & 4095;
  x32[i] = samples[b * 12288 + c * 4096 + p] * 2.f - 1.f;
}

// Masked + K-compacted f16 weights, TAP-MAJOR: wk16[layer][co][t*160 + cin].
// taps t: (-1,-1)(-1,0)(-1,1)(0,-1)(0,0)   (mask B, center kept)
// w1_16: 192x160 (rows >=160 zero), w2_16: 128x160 (rows >=100 zero)
__global__ void prep_w_kernel(const float* __restrict__ w_blk,
                              const float* __restrict__ w1,
                              const float* __restrict__ w2,
                              _Float16* __restrict__ wk16,
                              _Float16* __restrict__ w1_16,
                              _Float16* __restrict__ w2_16) {
  int i = blockIdx.x * 256 + threadIdx.x;
  if (i < 5 * 320 * KBLK) {
    int k = i % KBLK;
    int tmp = i / KBLK;
    int co = tmp % 320;
    int layer = tmp / 320;
    int t = k / 160;          // tap index 0..4
    int cin = k - t * 160;
    int ty = (t < 3) ? 0 : 1;
    int tx = (t < 3) ? t : (t - 3);
    wk16[i] = (_Float16)w_blk[(((layer * 320 + co) * 160 + cin) * 9) + ty * 3 + tx];
  }
  if (i < 192 * 160) {
    int co = i / 160, ci = i % 160;
    w1_16[i] = (_Float16)((co < 160) ? w1[co * 160 + ci] : 0.f);
  }
  if (i < 128 * 160) {
    int co = i / 160, ci = i % 160;
    w2_16[i] = (_Float16)((co < 100) ? w2[co * 160 + ci] : 0.f);
  }
}

// ---------------------------------------------------------------- 7x7 masked input conv (3 ch, 24 live taps)
__global__ __launch_bounds__(256) void conv0_kernel(const float* __restrict__ x32,
                                                    const float* __restrict__ w_in,
                                                    float* __restrict__ h,
                                                    _Float16* __restrict__ act) {
  int n  = blockIdx.x * 256 + threadIdx.x;     // 0..NPIX-1
  int co = blockIdx.y;                          // 0..159
  int y  = (n >> 6) & 63;
  int x  = n & 63;
  float s = 0.f;
#pragma unroll
  for (int cin = 0; cin < 3; ++cin) {
    const float* w  = w_in + (co * 3 + cin) * 49;
    const float* xc = x32 + cin * NPIX + n;
#pragma unroll
    for (int t = 0; t < 24; ++t) {   // mask A: taps strictly before center
      int dy = t / 7 - 3;
      int dx = t % 7 - 3;
      int yy = y + dy, xx = x + dx;
      if ((unsigned)yy < 64u && (unsigned)xx < 64u)
        s += w[t] * xc[dy * 64 + dx];
    }
  }
  h[co * NPIX + n]   = s;
  act[co * NPIX + n] = (_Float16)fmaxf(s, 0.f);
}

// ---------------------------------------------------------------- fused gated residual block
// Implicit-GEMM masked 3x3 conv (K=800, tap-major) + gate + residual + relu->f16.
// Double-buffered LDS tiles, async A fill, one barrier per K step.
__global__ __launch_bounds__(256) void resblock_kernel(const _Float16* __restrict__ act_in,
                                                       const _Float16* __restrict__ wk,
                                                       float* __restrict__ h,
                                                       _Float16* __restrict__ act_out) {
  const int tid  = threadIdx.x;
  const int lane = tid & 31;
  const int wave = tid >> 5;
  const int wm = wave >> 2;          // 0..1
  const int wn = wave & 3;           // 0..3
  const int n0 = blockIdx.x * 128;
  const int cbase = blockIdx.y * 32; // a-channel base
  const int NT = KBLK / 32;          // 25

  __shared__ _Float16 As[2][64 * LDA];
  __shared__ _Float16 Bs[2][128 * LDB];
  __shared__ float    Ss[64 * 128];

  // A-tile fill: thread -> (row, 8-half K chunk); async 16B copy to LDS.
  const int ar  = tid >> 2;            // 0..63
  const int akq = (tid & 3) * 8;       // 0,8,16,24
  const int aco = (ar < 32) ? (cbase + ar) : (160 + cbase + (ar - 32));
  const _Float16* asrc = wk + (size_t)aco * KBLK + akq;
  const unsigned a_lds0 = (unsigned)(uintptr_t)(&As[0][0] + ar * LDA + akq);
  const unsigned a_ldsz = (unsigned)(64 * LDA * 2);   // byte stride between buffers

  // B-tile fill: thread -> (col, K-half). Tap-major K: whole chunk shares one tap.
  const int bn  = tid >> 1;            // col 0..127
  const int bkh = (tid & 1) * 16;      // K-half 0/16
  const int pix = n0 + bn;
  const int y0  = (pix >> 6) & 63;
  const int x0  = pix & 63;

  auto fill = [&](int ktf, int buf) {
    async_copy_b128(a_lds0 + (unsigned)buf * a_ldsz, asrc + ktf * 32);
    const int tap = ktf / 5;
    const int sub = ktf - tap * 5;
    const int dy = (tap < 3) ? -1 : 0;
    const int dx = (tap < 3) ? (tap - 1) : (tap - 4);
    const int yy = y0 + dy, xx = x0 + dx;
    const bool valid = ((unsigned)yy < 64u) && ((unsigned)xx < 64u);
    const int cin0 = sub * 32 + bkh;
    const _Float16* src = act_in + (size_t)cin0 * NPIX + (pix + dy * 64 + dx);
    v8h lo = {}, hi = {};
    if (valid) {
#pragma unroll
      for (int i = 0; i < 8; ++i) lo[i] = src[(size_t)i * NPIX];
#pragma unroll
      for (int i = 0; i < 8; ++i) hi[i] = src[(size_t)(i + 8) * NPIX];
    }
    _Float16* bdst = &Bs[buf][0] + bn * LDB + bkh;
    *(v8h*)bdst       = lo;
    *(v8h*)(bdst + 8) = hi;
  };

  v8f acc[2][2] = {};

  fill(0, 0);
  for (int kt = 0; kt < NT; ++kt) {
    const int cur = kt & 1;
    wait_async0();          // async fill of tile `cur` complete (issued last step)
    __syncthreads();        // tile `cur` visible to all waves; prev reads of `cur^1` done
    if (kt + 1 < NT) fill(kt + 1, cur ^ 1);

    const _Float16* Ac = &As[cur][0];
    const _Float16* Bc = &Bs[cur][0];
    v16h a0 = load_frag_A(Ac, lane, wm * 32);
    v16h a1 = load_frag_A(Ac, lane, wm * 32 + 16);
    v16h b0 = load_frag_B(Bc, lane, wn * 32);
    v16h b1 = load_frag_B(Bc, lane, wn * 32 + 16);
    acc[0][0] = wmma_f16(a0, b0, acc[0][0]);
    acc[0][1] = wmma_f16(a0, b1, acc[0][1]);
    acc[1][0] = wmma_f16(a1, b0, acc[1][0]);
    acc[1][1] = wmma_f16(a1, b1, acc[1][1]);
  }
  __syncthreads();

  // stage conv result into LDS (C/D layout: VGPR r -> M = r + 8*(lane>=16), N = lane&15)
  {
    int lh = lane >> 4, lc = lane & 15;
#pragma unroll
    for (int i = 0; i < 2; ++i)
#pragma unroll
      for (int j = 0; j < 2; ++j)
#pragma unroll
        for (int r = 0; r < 8; ++r) {
          int m = wm * 32 + i * 16 + r + 8 * lh;
          int n = wn * 32 + j * 16 + lc;
          Ss[m * 128 + n] = acc[i][j][r];
        }
  }
  __syncthreads();

  // gate + residual + relu->f16
  for (int e = tid; e < 32 * 128; e += 256) {
    int r = e >> 7, n = e & 127;
    float av = Ss[r * 128 + n];
    float bv = Ss[(r + 32) * 128 + n];
    int c = cbase + r;
    int px = n0 + n;
    float hv = h[c * NPIX + px] + tanhf(av) * sigmoidf_(bv);
    h[c * NPIX + px] = hv;
    act_out[c * NPIX + px] = (_Float16)fmaxf(hv, 0.f);
  }
}

// ---------------------------------------------------------------- plain 1x1-conv GEMM (double-buffered)
__global__ __launch_bounds__(256) void gemm_kernel(const _Float16* __restrict__ A,     // [Mpad][K]
                                                   const _Float16* __restrict__ Bact,  // [K][NPIX]
                                                   float* __restrict__ out32,          // nullable
                                                   _Float16* __restrict__ out16relu,   // nullable
                                                   int K, int Mreal) {
  const int tid  = threadIdx.x;
  const int lane = tid & 31;
  const int wave = tid >> 5;
  const int wm = wave >> 2;
  const int wn = wave & 3;
  const int n0 = blockIdx.x * 128;
  const int m0 = blockIdx.y * 64;

  __shared__ _Float16 As[2][64 * LDA];
  __shared__ _Float16 Bs[2][128 * LDB];

  const int ar  = tid >> 2;
  const int akq = (tid & 3) * 8;
  const _Float16* asrc = A + (size_t)(m0 + ar) * K + akq;
  const unsigned a_lds0 = (unsigned)(uintptr_t)(&As[0][0] + ar * LDA + akq);
  const unsigned a_ldsz = (unsigned)(64 * LDA * 2);

  const int bn  = tid >> 1;
  const int bkh = (tid & 1) * 16;
  const _Float16* bsrc = Bact + n0 + bn;

  auto fill = [&](int ktf, int buf) {
    const int k0 = ktf * 32;
    async_copy_b128(a_lds0 + (unsigned)buf * a_ldsz, asrc + k0);
    v8h lo, hi;
#pragma unroll
    for (int i = 0; i < 8; ++i) lo[i] = bsrc[(size_t)(k0 + bkh + i) * NPIX];
#pragma unroll
    for (int i = 0; i < 8; ++i) hi[i] = bsrc[(size_t)(k0 + bkh + 8 + i) * NPIX];
    _Float16* bdst = &Bs[buf][0] + bn * LDB + bkh;
    *(v8h*)bdst       = lo;
    *(v8h*)(bdst + 8) = hi;
  };

  v8f acc[2][2] = {};
  const int NT = K / 32;

  fill(0, 0);
  for (int kt = 0; kt < NT; ++kt) {
    const int cur = kt & 1;
    wait_async0();
    __syncthreads();
    if (kt + 1 < NT) fill(kt + 1, cur ^ 1);

    const _Float16* Ac = &As[cur][0];
    const _Float16* Bc = &Bs[cur][0];
    v16h a0 = load_frag_A(Ac, lane, wm * 32);
    v16h a1 = load_frag_A(Ac, lane, wm * 32 + 16);
    v16h b0 = load_frag_B(Bc, lane, wn * 32);
    v16h b1 = load_frag_B(Bc, lane, wn * 32 + 16);
    acc[0][0] = wmma_f16(a0, b0, acc[0][0]);
    acc[0][1] = wmma_f16(a0, b1, acc[0][1]);
    acc[1][0] = wmma_f16(a1, b0, acc[1][0]);
    acc[1][1] = wmma_f16(a1, b1, acc[1][1]);
  }

  int lh = lane >> 4, lc = lane & 15;
#pragma unroll
  for (int i = 0; i < 2; ++i)
#pragma unroll
    for (int j = 0; j < 2; ++j)
#pragma unroll
      for (int r = 0; r < 8; ++r) {
        int co  = m0 + wm * 32 + i * 16 + r + 8 * lh;
        int px  = n0 + wn * 32 + j * 16 + lc;
        if (co < Mreal) {
          float v = acc[i][j][r];
          if (out32)     out32[co * NPIX + px] = v;
          if (out16relu) out16relu[co * NPIX + px] = (_Float16)fmaxf(v, 0.f);
        }
      }
}

// ---------------------------------------------------------------- DMOL log-prob + reduction
__global__ __launch_bounds__(256) void dmol_kernel(const float* __restrict__ params, // [100][NPIX]
                                                   const float* __restrict__ x32,    // [3][NPIX]
                                                   float* __restrict__ out) {
  const int tid = threadIdx.x;
  const int pix = blockIdx.x * 256 + tid;
  const int b   = pix >> 12;          // uniform per block (4096 pixels per batch)

  float x0 = x32[pix];
  float x1 = x32[NPIX + pix];
  float x2 = x32[2 * NPIX + pix];

  float logit[10];
  float mx = -1e30f;
#pragma unroll
  for (int m = 0; m < 10; ++m) {
    logit[m] = params[m * NPIX + pix];
    mx = fmaxf(mx, logit[m]);
  }
  float se = 0.f;
#pragma unroll
  for (int m = 0; m < 10; ++m) se += __expf(logit[m] - mx);
  float lse_logits = mx + __logf(se);

  float lpmax = -1e30f;
  float lps[10];
#pragma unroll
  for (int m = 0; m < 10; ++m) {
    float mu0 = params[(10 + 0 * 30 + m) * NPIX + pix];
    float mu1 = params[(10 + 1 * 30 + m) * NPIX + pix];
    float mu2 = params[(10 + 2 * 30 + m) * NPIX + pix];
    float ls0 = fmaxf(params[(10 + 0 * 30 + 10 + m) * NPIX + pix], -7.f);
    float ls1 = fmaxf(params[(10 + 1 * 30 + 10 + m) * NPIX + pix], -7.f);
    float ls2 = fmaxf(params[(10 + 2 * 30 + 10 + m) * NPIX + pix], -7.f);
    float c0  = tanhf(params[(10 + 0 * 30 + 20 + m) * NPIX + pix]);
    float c1  = tanhf(params[(10 + 1 * 30 + 20 + m) * NPIX + pix]);
    float c2  = tanhf(params[(10 + 2 * 30 + 20 + m) * NPIX + pix]);

    float mean[3] = {mu0, mu1 + c0 * x0, mu2 + c1 * x0 + c2 * x1};
    float xs[3]   = {x0, x1, x2};
    float lsc[3]  = {ls0, ls1, ls2};

    float lp = logit[m] - lse_logits;
#pragma unroll
    for (int c = 0; c < 3; ++c) {
      float centered = xs[c] - mean[c];
      float inv = __expf(-lsc[c]);
      float plus_in = inv * (centered + 0.003921568627f);
      float min_in  = inv * (centered - 0.003921568627f);
      float cdf_delta = sigmoidf_(plus_in) - sigmoidf_(min_in);
      float log_cdf_plus = plus_in - softplusf_(plus_in);
      float log_om_cdf_min = -softplusf_(min_in);
      float mid_in = inv * centered;
      float log_pdf_mid = mid_in - lsc[c] - 2.f * softplusf_(mid_in);
      float log_prob_in = (cdf_delta > 1e-5f)
                              ? __logf(fmaxf(cdf_delta, 1e-12f))
                              : (log_pdf_mid - 4.852030263919617f); // log(127.5)
      float lpc = (xs[c] < -0.999f) ? log_cdf_plus
                 : ((xs[c] > 0.999f) ? log_om_cdf_min : log_prob_in);
      lp += lpc;
    }
    lps[m] = lp;
    lpmax = fmaxf(lpmax, lp);
  }
  float s = 0.f;
#pragma unroll
  for (int m = 0; m < 10; ++m) s += __expf(lps[m] - lpmax);
  float lp_pix = lpmax + __logf(s);

  __shared__ float red[256];
  red[tid] = lp_pix;
  __syncthreads();
  for (int st = 128; st > 0; st >>= 1) {
    if (tid < st) red[tid] += red[tid + st];
    __syncthreads();
  }
  if (tid == 0) atomicAdd(&out[b], red[0]);
}

// ---------------------------------------------------------------- launcher
extern "C" void kernel_launch(void* const* d_in, const int* in_sizes, int n_in,
                              void* d_out, int out_size, void* d_ws, size_t ws_size,
                              hipStream_t stream) {
  const float* samples = (const float*)d_in[0];
  const float* w_in    = (const float*)d_in[1];
  const float* w_blk   = (const float*)d_in[2];
  const float* w_out1  = (const float*)d_in[3];
  const float* w_out2  = (const float*)d_in[4];
  float* out = (float*)d_out;

  char* ws = (char*)d_ws;
  size_t off = 0;
  auto alloc = [&](size_t bytes) { void* p = ws + off; off += (bytes + 255) & ~size_t(255); return p; };

  float*    x32    = (float*)   alloc(3   * (size_t)NPIX * 4);
  float*    h      = (float*)   alloc(160 * (size_t)NPIX * 4);
  _Float16* actA   = (_Float16*)alloc(160 * (size_t)NPIX * 2);
  _Float16* actB   = (_Float16*)alloc(160 * (size_t)NPIX * 2);
  _Float16* wk16   = (_Float16*)alloc(5 * 320 * (size_t)KBLK * 2);
  _Float16* w1_16  = (_Float16*)alloc(192 * 160 * 2);
  _Float16* w2_16  = (_Float16*)alloc(128 * 160 * 2);
  float*    params = (float*)   alloc(100 * (size_t)NPIX * 4);
  (void)ws_size; (void)in_sizes; (void)n_in; (void)out_size;

  zero_out_kernel<<<1, 32, 0, stream>>>(out);
  prep_x_kernel<<<(3 * NPIX + 255) / 256, 256, 0, stream>>>(samples, x32);
  prep_w_kernel<<<(5 * 320 * KBLK + 255) / 256, 256, 0, stream>>>(w_blk, w_out1, w_out2,
                                                                  wk16, w1_16, w2_16);
  conv0_kernel<<<dim3(NPIX / 256, 160), 256, 0, stream>>>(x32, w_in, h, actA);

  _Float16* ain = actA;
  _Float16* aout = actB;
  for (int i = 0; i < 5; ++i) {
    resblock_kernel<<<dim3(NPIX / 128, 5), 256, 0, stream>>>(ain, wk16 + (size_t)i * 320 * KBLK,
                                                             h, aout);
    _Float16* t = ain; ain = aout; aout = t;
  }
  // ain == relu(h5);  h_out = relu(W1 * ain)  -> aout (f16)
  gemm_kernel<<<dim3(NPIX / 128, 3), 256, 0, stream>>>(w1_16, ain, nullptr, aout, 160, 160);
  // params = W2 * h_out
  gemm_kernel<<<dim3(NPIX / 128, 2), 256, 0, stream>>>(w2_16, aout, params, nullptr, 160, 100);

  dmol_kernel<<<NPIX / 256, 256, 0, stream>>>(params, x32, out);
}
